// GeometryHeadSparse_26121991094530
// MI455X (gfx1250) — matrix operand order repaired and checked by reading, not compiled, not run on hardware
//
#include <hip/hip_runtime.h>
#include <hip/hip_bf16.h>

// ---- fixed problem sizes from the reference ----
constexpr int N    = 120000;   // total points
constexpr int NPB  = 60000;    // points per batch (B=2, contiguous)
constexpr int C    = 64;       // channels
constexpr int K27  = 27;       // kernel offsets
constexpr float EPS = 1e-5f;
constexpr int WSET  = K27 * C * C;       // 110592 elems per 64x64 weight set
constexpr int TILES32 = N / 32;          // 3750 row-tiles of 32
constexpr int LDS_STRIDE = 72;           // halves per weight row (pad 64->72: 16B aligned, bank-spread)
constexpr size_t LDS_BYTES = (size_t)K27 * 64 * LDS_STRIDE * 2;  // 248832 B < 320KB WGP LDS

typedef __attribute__((ext_vector_type(16))) _Float16 v16h;
typedef __attribute__((ext_vector_type(8)))  _Float16 v8h;
typedef __attribute__((ext_vector_type(8)))  float    v8f;

union Frag16 { v16h v; v8h h[2]; };

// ---------------------------------------------------------------------------
// Weight transpose+convert: w[k][cin][cout] f32  ->  wT[k][cout][cin] f16
// ---------------------------------------------------------------------------
__global__ __launch_bounds__(256) void convert_w_kernel(
    const float* __restrict__ w, _Float16* __restrict__ wT)
{
  int set = blockIdx.y;
  int e = blockIdx.x * 256 + threadIdx.x;
  if (e >= WSET) return;
  int k  = e >> 12;          // /4096
  int n  = (e >> 6) & 63;    // dst row  = cout
  int kk = e & 63;           // dst col  = cin (K)
  wT[(size_t)set * WSET + e] =
      (_Float16)w[(size_t)set * WSET + k * 4096 + kk * 64 + n];
}

// ---------------------------------------------------------------------------
// Inverse kernel-map: inv[k][o] = input row feeding output o (sentinel = N)
// ---------------------------------------------------------------------------
__global__ __launch_bounds__(256) void inv_init_kernel(int* __restrict__ inv)
{
  int i = blockIdx.x * 256 + threadIdx.x;
  if (i >= N) return;
  inv[(size_t)blockIdx.y * N + i] = N;
}

__global__ __launch_bounds__(256) void inv_scatter_kernel(
    const int* __restrict__ in_idx, const int* __restrict__ out_idx,
    int* __restrict__ inv)
{
  int i = blockIdx.x * 256 + threadIdx.x;
  if (i >= N) return;
  size_t e = (size_t)blockIdx.y * N + i;
  int o = out_idx[e];
  if (o < N) inv[(size_t)blockIdx.y * N + o] = in_idx[e];
}

// ---------------------------------------------------------------------------
// Instance-norm statistics: per (batch, channel) sum / sumsq
// stats layout: [0..127]=sum(b*64+c), [128..255]=sumsq
// ---------------------------------------------------------------------------
__global__ __launch_bounds__(256) void zero_stats_kernel(float* __restrict__ s)
{
  s[threadIdx.x] = 0.f;
}

__global__ __launch_bounds__(256) void reduce_stats_kernel(
    const float* __restrict__ f, float* __restrict__ stats)
{
  int c   = threadIdx.x & 63;
  int sub = threadIdx.x >> 6;            // 0..3
  int base = blockIdx.x * 1024;
  float s0 = 0.f, s1 = 0.f, q0 = 0.f, q1 = 0.f;
  for (int i = 0; i < 256; ++i) {
    int r = base + i * 4 + sub;
    if (r < N) {
      float v = f[(size_t)r * 64 + c];
      if (r < NPB) { s0 += v; q0 += v * v; }
      else         { s1 += v; q1 += v * v; }
    }
  }
  __shared__ float red[4][4][64];
  red[sub][0][c] = s0; red[sub][1][c] = s1;
  red[sub][2][c] = q0; red[sub][3][c] = q1;
  __syncthreads();
  if (sub == 0) {
#pragma unroll
    for (int v = 0; v < 4; ++v) {
      float t = red[0][v][c] + red[1][v][c] + red[2][v][c] + red[3][v][c];
      atomicAdd(&stats[v * 64 + c], t);
    }
  }
}

__global__ __launch_bounds__(128) void finalize_stats_kernel(
    const float* __restrict__ stats, float* __restrict__ meanv,
    float* __restrict__ istdv)
{
  int i = threadIdx.x;                   // b*64+c, 128 threads
  float mean = stats[i] / (float)NPB;
  float var  = stats[128 + i] / (float)NPB - mean * mean;
  meanv[i] = mean;
  istdv[i] = rsqrtf(var + EPS);
}

// ---------------------------------------------------------------------------
// Normalize (+ optional residual) + ReLU; writes optional f32 and f16 copies.
// Covers rows 0..N: row N is the zero sentinel row of the f16 buffer.
// ---------------------------------------------------------------------------
__global__ __launch_bounds__(256) void norm_act_kernel(
    const float* __restrict__ f, const float* __restrict__ meanv,
    const float* __restrict__ istdv, const float* __restrict__ res,
    float* __restrict__ out32, _Float16* __restrict__ outh)
{
  size_t e = (size_t)blockIdx.x * 256 + threadIdx.x;
  if (e >= (size_t)(N + 1) * 64) return;
  int r = (int)(e >> 6);
  int c = (int)(e & 63);
  if (r == N) { outh[e] = (_Float16)0.f; return; }
  int b = (r >= NPB) ? 1 : 0;
  float v = (f[e] - meanv[b * 64 + c]) * istdv[b * 64 + c];
  if (res)  v += res[e];
  v = v > 0.f ? v : 0.f;
  if (out32) out32[e] = v;
  outh[e] = (_Float16)v;
}

// ---------------------------------------------------------------------------
// WMMA sparse conv 64->64.
//  * 512-thread blocks (16 waves). All 27 weight matrices (249KB f16, padded
//    row stride 72 halves) are cooperatively staged into LDS once; the k-loop
//    reads B fragments via ds_load_b128 with no barriers.
//  * Each wave owns a 32-row output tile (two stacked 16-row WMMA tiles that
//    share B fragments), accumulating all 27 offsets in 8x v8f accumulators.
//   A (16x32 f16): lane l = row l&15; halves = K {hi*8.., 16+hi*8..}
//   B (32x16 f16): lane l = col l&15; halves = K {hi*8.., 16+hi*8..}
//   C (16x16 f32): lane l = col l&15; vgpr p = row p + 8*hi
// ---------------------------------------------------------------------------
__global__ __launch_bounds__(512) void conv64_wmma_kernel(
    const _Float16* __restrict__ xh,   // (N+1) x 64, row N = zeros
    const _Float16* __restrict__ wT,   // 27 x 64(n) x 64(K)
    const int* __restrict__ inv,       // 27 x N
    float* __restrict__ out)           // N x 64
{
  extern __shared__ _Float16 wlds[];   // [27*64 rows][72 halves]
  const int tid = threadIdx.x;

  // cooperative preload: 27*64 rows, 8 chunks of 8 halves each
  for (int i = tid; i < K27 * 64 * 8; i += 512) {
    int row = i >> 3;
    int off = (i & 7) * 8;
    *(v8h*)(wlds + (size_t)row * LDS_STRIDE + off) =
        *(const v8h*)(wT + (size_t)row * 64 + off);
  }
  __syncthreads();

  const int lane = tid & 31;
  const int wave = tid >> 5;
  const int t32  = blockIdx.x * 16 + wave;
  if (t32 >= TILES32) return;                // wave-uniform exit (after barrier)
  const int row0 = t32 * 32;
  const int m  = lane & 15;
  const int hi = lane >> 4;

  v8f acc[2][4];
#pragma unroll
  for (int u = 0; u < 2; ++u)
#pragma unroll
    for (int t = 0; t < 4; ++t) acc[u][t] = (v8f)0.f;

  const int o0 = row0 + m;
  const int o1 = row0 + 16 + m;
  for (int k = 0; k < K27; ++k) {
    const int i0 = inv[(size_t)k * N + o0];          // 0..N (N => zero row)
    const int i1 = inv[(size_t)k * N + o1];
    const _Float16* r0 = xh + (size_t)i0 * 64;
    const _Float16* r1 = xh + (size_t)i1 * 64;
    Frag16 a[2][2];
    a[0][0].h[0] = *(const v8h*)(r0 + hi * 8);       // tile0, K  0..31
    a[0][0].h[1] = *(const v8h*)(r0 + 16 + hi * 8);
    a[0][1].h[0] = *(const v8h*)(r0 + 32 + hi * 8);  // tile0, K 32..63
    a[0][1].h[1] = *(const v8h*)(r0 + 48 + hi * 8);
    a[1][0].h[0] = *(const v8h*)(r1 + hi * 8);       // tile1, K  0..31
    a[1][0].h[1] = *(const v8h*)(r1 + 16 + hi * 8);
    a[1][1].h[0] = *(const v8h*)(r1 + 32 + hi * 8);  // tile1, K 32..63
    a[1][1].h[1] = *(const v8h*)(r1 + 48 + hi * 8);

    const _Float16* wk = wlds + (size_t)k * 64 * LDS_STRIDE;
#pragma unroll
    for (int t = 0; t < 4; ++t) {
      const _Float16* wp = wk + (size_t)(t * 16 + m) * LDS_STRIDE;
      Frag16 b0, b1;
      b0.h[0] = *(const v8h*)(wp + hi * 8);
      b0.h[1] = *(const v8h*)(wp + 16 + hi * 8);
      b1.h[0] = *(const v8h*)(wp + 32 + hi * 8);
      b1.h[1] = *(const v8h*)(wp + 48 + hi * 8);
      acc[0][t] = __builtin_amdgcn_wmma_f32_16x16x32_f16(
          false, a[0][0].v, false, b0.v, (short)0, acc[0][t], false, false);
      acc[0][t] = __builtin_amdgcn_wmma_f32_16x16x32_f16(
          false, a[0][1].v, false, b1.v, (short)0, acc[0][t], false, false);
      acc[1][t] = __builtin_amdgcn_wmma_f32_16x16x32_f16(
          false, a[1][0].v, false, b0.v, (short)0, acc[1][t], false, false);
      acc[1][t] = __builtin_amdgcn_wmma_f32_16x16x32_f16(
          false, a[1][1].v, false, b1.v, (short)0, acc[1][t], false, false);
    }
  }

  // plain coalesced stores: rows are contiguous, no atomics needed
#pragma unroll
  for (int u = 0; u < 2; ++u) {
#pragma unroll
    for (int t = 0; t < 4; ++t) {
#pragma unroll
      for (int p = 0; p < 8; ++p) {
        int mr = p + 8 * hi;
        out[(size_t)(row0 + u * 16 + mr) * 64 + t * 16 + m] = acc[u][t][p];
      }
    }
  }
}

// ---------------------------------------------------------------------------
// Final conv 64->1 (+bias): gather-dot per output row (tiny FLOP count)
// ---------------------------------------------------------------------------
__global__ __launch_bounds__(256) void final_conv_kernel(
    const _Float16* __restrict__ xh, const int* __restrict__ inv,
    const float* __restrict__ wf,    // 27 x 64
    const float* __restrict__ bias, float* __restrict__ out)
{
  int r = blockIdx.x * 256 + threadIdx.x;
  if (r >= N) return;
  float acc = bias[0];
  for (int k = 0; k < K27; ++k) {
    const int iin = inv[(size_t)k * N + r];
    const _Float16* row = xh + (size_t)iin * 64;
    const float* w = wf + k * 64;
    float a = 0.f;
#pragma unroll
    for (int c = 0; c < 64; ++c) a += (float)row[c] * w[c];
    acc += a;
  }
  out[r] = acc;
}

// ---------------------------------------------------------------------------
extern "C" void kernel_launch(void* const* d_in, const int* in_sizes, int n_in,
                              void* d_out, int out_size, void* d_ws, size_t ws_size,
                              hipStream_t stream) {
  const float* feats    = (const float*)d_in[0];
  const float* w_blocks = (const float*)d_in[1];   // [2][2][27][64][64]
  const float* w_final  = (const float*)d_in[2];   // [27][64][1]
  const float* b_final  = (const float*)d_in[3];   // [1]
  const int*   in_idx   = (const int*)d_in[4];     // [27][N]
  const int*   out_idx  = (const int*)d_in[5];     // [27][N]
  float* out = (float*)d_out;

  // workspace carving (256B aligned)
  char* p = (char*)d_ws;
  auto take = [&](size_t bytes) { char* r = p; p += (bytes + 255) & ~(size_t)255; return r; };
  float*    x32   = (float*)take((size_t)N * 64 * 4);
  float*    h32   = (float*)take((size_t)N * 64 * 4);
  _Float16* xh    = (_Float16*)take((size_t)(N + 1) * 64 * 2);
  _Float16* yh    = (_Float16*)take((size_t)(N + 1) * 64 * 2);
  int*      inv   = (int*)take((size_t)K27 * N * 4);
  _Float16* wT    = (_Float16*)take((size_t)4 * WSET * 2);
  float*    stats = (float*)take(256 * 4);
  float*    meanv = (float*)take(128 * 4);
  float*    istdv = (float*)take(128 * 4);

  const dim3 blk(256);
  const dim3 gW((WSET + 255) / 256, 4);
  const dim3 gInv((N + 255) / 256, K27);
  const int  gRed  = (N + 1023) / 1024;
  const int  gNorm = (int)(((size_t)(N + 1) * 64 + 255) / 256);
  const int  gConv = (TILES32 + 15) / 16;           // 16 waves/block, 32 rows/wave
  const int  gRow  = (N + 255) / 256;

  // one-time-per-call prep (deterministic)
  convert_w_kernel<<<gW, blk, 0, stream>>>(w_blocks, wT);
  inv_init_kernel<<<gInv, blk, 0, stream>>>(inv);
  inv_scatter_kernel<<<gInv, blk, 0, stream>>>(in_idx, out_idx, inv);

  // x = relu(IN(feats)); write f32 + f16 (with zero sentinel row)
  zero_stats_kernel<<<1, blk, 0, stream>>>(stats);
  reduce_stats_kernel<<<gRed, blk, 0, stream>>>(feats, stats);
  finalize_stats_kernel<<<1, 128, 0, stream>>>(stats, meanv, istdv);
  norm_act_kernel<<<gNorm, blk, 0, stream>>>(feats, meanv, istdv, nullptr, x32, xh);

  for (int b = 0; b < 2; ++b) {
    // h = conv(x); y = relu(IN(h))  (f16 only)
    conv64_wmma_kernel<<<gConv, 512, LDS_BYTES, stream>>>(xh, wT + (size_t)(2 * b) * WSET, inv, h32);
    zero_stats_kernel<<<1, blk, 0, stream>>>(stats);
    reduce_stats_kernel<<<gRed, blk, 0, stream>>>(h32, stats);
    finalize_stats_kernel<<<1, 128, 0, stream>>>(stats, meanv, istdv);
    norm_act_kernel<<<gNorm, blk, 0, stream>>>(h32, meanv, istdv, nullptr, nullptr, yh);

    // h = conv(y); x = relu(IN(h) + x)  (f32 in-place + f16)
    conv64_wmma_kernel<<<gConv, 512, LDS_BYTES, stream>>>(yh, wT + (size_t)(2 * b + 1) * WSET, inv, h32);
    zero_stats_kernel<<<1, blk, 0, stream>>>(stats);
    reduce_stats_kernel<<<gRed, blk, 0, stream>>>(h32, stats);
    finalize_stats_kernel<<<1, 128, 0, stream>>>(stats, meanv, istdv);
    norm_act_kernel<<<gNorm, blk, 0, stream>>>(h32, meanv, istdv, x32, x32, xh);
  }

  final_conv_kernel<<<gRow, blk, 0, stream>>>(xh, inv, w_final, b_final, out);
}
